// PointNetMsg_55688545960044
// MI455X (gfx1250) — matrix operand order repaired and checked by reading, not compiled, not run on hardware
//
#include <hip/hip_runtime.h>
#include <hip/hip_bf16.h>
#include <math.h>

typedef __attribute__((ext_vector_type(16))) _Float16 v16h;
typedef __attribute__((ext_vector_type(8)))  _Float16 v8h;
typedef __attribute__((ext_vector_type(8)))  float    v8f;

static __host__ __device__ inline int round32(int x) { return (x + 31) & ~31; }

// ---------------------------------------------------------------------------
// GEMM with WMMA:  out = act(A[M x K] * W[K x N] + bias[N])
//  - A: f16, row stride lda = round32(K); rows beyond M readable (padded bufs)
//  - Wp: f16 packed in B-fragment order [kt][nt][lane][16], zero-padded so no
//        bounds checks are needed in the K loop (A pad * 0 == 0).
//  - One wave per 16x16 D tile; block = 8 waves = 128 rows; grid.y = Nt.
//  - Two independent accumulator chains (K split even/odd) so consecutive
//    WMMAs have no D->C dependency and can interleave.
// ---------------------------------------------------------------------------
template <bool RELU, bool OUT_F32>
__global__ void gemm_wmma(const _Float16* __restrict__ A, int lda,
                          const _Float16* __restrict__ Wp,
                          const float* __restrict__ bias,
                          void* __restrict__ out,
                          int M, int Kt, int Nt, int N, int ldo) {
  const int lane = threadIdx.x & 31;
  const int wave = threadIdx.x >> 5;
  const int m0 = blockIdx.x * 128 + wave * 16;
  if (m0 >= M) return;  // wave-uniform
  const int nt = blockIdx.y;
  const int hg = lane >> 4;  // half-group
  const int lr = lane & 15;

  const _Float16* Ar = A + (size_t)(m0 + lr) * lda + hg * 8;
  const _Float16* Wr = Wp + ((size_t)nt * 32 + lane) * 16;
  const size_t wstep = (size_t)Nt * 32 * 16;  // halves per k-tile of packed W

  v8f acc0 = {}, acc1 = {};
  int kt = 0;
  for (; kt + 2 <= Kt; kt += 2) {
    __builtin_prefetch((const void*)(Ar + (size_t)kt * 32 + 512), 0, 1);
    v8h a0lo = *(const v8h*)(Ar + (size_t)kt * 32);
    v8h a0hi = *(const v8h*)(Ar + (size_t)kt * 32 + 16);
    v8h a1lo = *(const v8h*)(Ar + (size_t)(kt + 1) * 32);
    v8h a1hi = *(const v8h*)(Ar + (size_t)(kt + 1) * 32 + 16);
    v16h b0 = *(const v16h*)(Wr + (size_t)kt * wstep);
    v16h b1 = *(const v16h*)(Wr + (size_t)(kt + 1) * wstep);
    v16h a0 = __builtin_shufflevector(a0lo, a0hi, 0, 1, 2, 3, 4, 5, 6, 7,
                                      8, 9, 10, 11, 12, 13, 14, 15);
    v16h a1 = __builtin_shufflevector(a1lo, a1hi, 0, 1, 2, 3, 4, 5, 6, 7,
                                      8, 9, 10, 11, 12, 13, 14, 15);
    acc0 = __builtin_amdgcn_wmma_f32_16x16x32_f16(false, a0, false, b0,
                                                  (short)0, acc0, false, false);
    acc1 = __builtin_amdgcn_wmma_f32_16x16x32_f16(false, a1, false, b1,
                                                  (short)0, acc1, false, false);
  }
  if (kt < Kt) {
    v8h alo = *(const v8h*)(Ar + (size_t)kt * 32);
    v8h ahi = *(const v8h*)(Ar + (size_t)kt * 32 + 16);
    v16h a = __builtin_shufflevector(alo, ahi, 0, 1, 2, 3, 4, 5, 6, 7,
                                     8, 9, 10, 11, 12, 13, 14, 15);
    v16h b = *(const v16h*)(Wr + (size_t)kt * wstep);
    acc0 = __builtin_amdgcn_wmma_f32_16x16x32_f16(false, a, false, b,
                                                  (short)0, acc0, false, false);
  }

  const int col = nt * 16 + lr;
  const float bv = (bias && col < N) ? bias[col] : 0.0f;
#pragma unroll
  for (int v = 0; v < 8; ++v) {
    int row = m0 + v + hg * 8;
    if (row < M) {
      float r = (col < N) ? (acc0[v] + acc1[v] + bv) : 0.0f;
      if (RELU) r = fmaxf(r, 0.0f);
      if (OUT_F32) ((float*)out)[(size_t)row * ldo + col] = r;
      else ((_Float16*)out)[(size_t)row * ldo + col] = (_Float16)r;
    }
  }
}

// ---------------------------------------------------------------------------
// Pack fp32 weights [K x N] -> f16 fragment layout [kt][nt][lane][16], padded
// with zeros (K pad rows and N pad cols).
// ---------------------------------------------------------------------------
__global__ void pack_weights(const float* __restrict__ W, int K, int N,
                             int Kt, int Nt, _Float16* __restrict__ out) {
  int total = Kt * Nt * 512;
  int tid = blockIdx.x * blockDim.x + threadIdx.x;
  if (tid >= total) return;
  int h = tid & 15;
  int lane = (tid >> 4) & 31;
  int t = tid >> 9;
  int nt = t % Nt;
  int kt = t / Nt;
  int hg = lane >> 4, lr = lane & 15;
  int kk = kt * 32 + hg * 8 + (h < 8 ? h : h + 8);
  int nn = nt * 16 + lr;
  out[tid] = (kk < K && nn < N) ? (_Float16)W[(size_t)kk * N + nn] : (_Float16)0.0f;
}

// ---------------------------------------------------------------------------
// Farthest point sampling: one block per batch.  Entire xyz (up to 96 KB) and
// the running dist array live in LDS (MI455X: 320 KB per WGP) so the 358
// sequential iterations touch no global memory.  Lowest-index tie-break
// matches jnp.argmax.
// Dynamic LDS: dist[N] + sx[N] + sy[N] + sz[N] + rmax[bd] + rarg[bd] + sfar
// ---------------------------------------------------------------------------
__global__ void fps_kernel(const float* __restrict__ xyz, int N, int npoint,
                           int* __restrict__ out_idx) {
  extern __shared__ float smem[];
  float* dist = smem;
  float* sx = smem + N;
  float* sy = sx + N;
  float* sz = sy + N;
  float* rmax = sz + N;
  int* rarg = (int*)(rmax + blockDim.x);
  int* sfar = rarg + blockDim.x;
  const int b = blockIdx.x;
  const float* X = xyz + (size_t)b * N * 3;
  for (int i = threadIdx.x; i < N; i += blockDim.x) {
    dist[i] = 1e10f;
    sx[i] = X[i * 3 + 0];
    sy[i] = X[i * 3 + 1];
    sz[i] = X[i * 3 + 2];
  }
  if (threadIdx.x == 0) *sfar = 0;
  __syncthreads();
  for (int it = 0; it < npoint; ++it) {
    const int far = *sfar;
    if (threadIdx.x == 0) out_idx[b * npoint + it] = far;
    const float cx = sx[far], cy = sy[far], cz = sz[far];
    float bm = -1.0f; int ba = 0;
    for (int i = threadIdx.x; i < N; i += blockDim.x) {
      float dx = sx[i] - cx, dy = sy[i] - cy, dz = sz[i] - cz;
      float d = dx * dx + dy * dy + dz * dz;
      float nd = fminf(dist[i], d);
      dist[i] = nd;
      if (nd > bm) { bm = nd; ba = i; }
    }
    rmax[threadIdx.x] = bm; rarg[threadIdx.x] = ba;
    __syncthreads();
    for (int s = blockDim.x >> 1; s > 0; s >>= 1) {
      if ((int)threadIdx.x < s) {
        float o = rmax[threadIdx.x + s]; int oa = rarg[threadIdx.x + s];
        if (o > rmax[threadIdx.x] || (o == rmax[threadIdx.x] && oa < rarg[threadIdx.x])) {
          rmax[threadIdx.x] = o; rarg[threadIdx.x] = oa;
        }
      }
      __syncthreads();
    }
    if (threadIdx.x == 0) *sfar = rarg[0];
    __syncthreads();
  }
}

__global__ void gather_xyz(const float* __restrict__ xyz, const int* __restrict__ idx,
                           int N, int S, int total /*B*S*/, float* __restrict__ out) {
  int tid = blockIdx.x * blockDim.x + threadIdx.x;
  if (tid >= total * 3) return;
  int row = tid / 3, k = tid % 3, b = row / S;
  out[tid] = xyz[((size_t)(b * N + idx[row])) * 3 + k];
}

// ---------------------------------------------------------------------------
// Ball query: one wave per center; ballot + prefix-popc appends qualifying
// indices in ascending order (== reference's sort of masked arange).
// ---------------------------------------------------------------------------
__global__ void query_ball(const float* __restrict__ xyz, const float* __restrict__ new_xyz,
                           int N, int S, float r2, int ns, int* __restrict__ gi,
                           int total /*B*S*/) {
  const int w = blockIdx.x * (blockDim.x >> 5) + (threadIdx.x >> 5);
  if (w >= total) return;
  const int lane = threadIdx.x & 31;
  const int b = w / S;
  const float cx = new_xyz[w * 3 + 0], cy = new_xyz[w * 3 + 1], cz = new_xyz[w * 3 + 2];
  const float* X = xyz + (size_t)b * N * 3;
  int* out = gi + (size_t)w * ns;
  int count = 0, first = -1;
  for (int n0 = 0; n0 < N && count < ns; n0 += 32) {
    int n = n0 + lane;
    bool pred = false;
    if (n < N) {
      float dx = X[n * 3 + 0] - cx, dy = X[n * 3 + 1] - cy, dz = X[n * 3 + 2] - cz;
      pred = (dx * dx + dy * dy + dz * dz) <= r2;
    }
    unsigned mask = (unsigned)__ballot(pred);
    if (first < 0 && mask) first = n0 + __ffs(mask) - 1;
    int before = __popc(mask & ((1u << lane) - 1u));
    int pos = count + before;
    if (pred && pos < ns) out[pos] = n;
    count += __popc(mask);
  }
  if (count > ns) count = ns;
  if (first < 0) first = 0;
  for (int j = count + lane; j < ns; j += 32) out[j] = first;
}

// ---------------------------------------------------------------------------
// Grouped features, one thread per OUTPUT ELEMENT (coalesced stores):
// out[row, 0..Cp-1] = points[b,g], out[row, Cp..Cp+2] = xyz[b,g]-new_xyz[b,s],
// zeros to Cpad.
// ---------------------------------------------------------------------------
template <typename PT>
__global__ void group_points(const float* __restrict__ xyz, const float* __restrict__ new_xyz,
                             const PT* __restrict__ points, int Cp,
                             const int* __restrict__ gi, int N, int S, int ns,
                             int totalRows, int Cpad, _Float16* __restrict__ out) {
  size_t tid = (size_t)blockIdx.x * blockDim.x + threadIdx.x;
  size_t totalE = (size_t)totalRows * Cpad;
  if (tid >= totalE) return;
  int row = (int)(tid / Cpad);
  int c = (int)(tid % Cpad);
  int bs = row / ns;
  int b = bs / S;
  _Float16 v = (_Float16)0.0f;
  if (c < Cp + 3) {
    int g = gi[row];
    if (g >= N) g = N - 1;
    if (c < Cp) v = (_Float16)(float)points[(size_t)(b * N + g) * Cp + c];
    else v = (_Float16)(xyz[(size_t)(b * N + g) * 3 + (c - Cp)] -
                        new_xyz[(size_t)bs * 3 + (c - Cp)]);
  }
  out[tid] = v;
}

// out[g, coff+c] = max_j in[(g*ns+j), c]  (in has row stride C, C mult of 32)
__global__ void maxpool_rows(const _Float16* __restrict__ in, int G, int ns, int C,
                             _Float16* __restrict__ out, int outC, int coff) {
  int tid = blockIdx.x * blockDim.x + threadIdx.x;
  if (tid >= G * C) return;
  int g = tid / C, c = tid % C;
  const _Float16* p = in + (size_t)g * ns * C + c;
  float m = -1e30f;
  for (int j = 0; j < ns; ++j) m = fmaxf(m, (float)p[(size_t)j * C]);
  out[(size_t)g * outC + coff + c] = (_Float16)m;
}

// group_all pack: channels = 3 (xyz f32) + Cp (points f16), zero pad to Cpad
__global__ void pack_groupall(const float* __restrict__ xyz, const _Float16* __restrict__ pts,
                              int Cp, int total /*B*S*/, int Cpad, _Float16* __restrict__ out) {
  int tid = blockIdx.x * blockDim.x + threadIdx.x;
  if (tid >= total * Cpad) return;
  int row = tid / Cpad, c = tid % Cpad;
  _Float16 v = (_Float16)0.0f;
  if (c < 3) v = (_Float16)xyz[(size_t)row * 3 + c];
  else if (c < Cp + 3) v = pts[(size_t)row * Cp + (c - 3)];
  out[tid] = v;
}

// concat [p1 (C1) | p2 (C2)] -> f16 rows with stride Cpad (zero pad)
__global__ void pack_concat_f16(const _Float16* __restrict__ p1, int C1,
                                const _Float16* __restrict__ p2, int C2,
                                int total /*rows*/, int Cpad, _Float16* __restrict__ out) {
  int tid = blockIdx.x * blockDim.x + threadIdx.x;
  if (tid >= total * Cpad) return;
  int row = tid / Cpad, c = tid % Cpad;
  _Float16 v = (_Float16)0.0f;
  if (c < C1) v = p1[(size_t)row * C1 + c];
  else if (c < C1 + C2) v = p2[(size_t)row * C2 + (c - C1)];
  out[tid] = v;
}

// up1 pack: [x(3) | x(3) | interp(128)] = 134 channels, padded to 160
__global__ void pack_up1(const float* __restrict__ x, const _Float16* __restrict__ interp,
                         int total /*B*8192*/, _Float16* __restrict__ out) {
  const int Cpad = 160;
  int tid = blockIdx.x * blockDim.x + threadIdx.x;
  if (tid >= total * Cpad) return;
  int row = tid / Cpad, c = tid % Cpad;
  _Float16 v = (_Float16)0.0f;
  if (c < 3)        v = (_Float16)x[(size_t)row * 3 + c];
  else if (c < 6)   v = (_Float16)x[(size_t)row * 3 + (c - 3)];
  else if (c < 134) v = interp[(size_t)row * 128 + (c - 6)];
  out[tid] = v;
}

// broadcast src[b, :] over all S points of batch b
__global__ void broadcast_pts(const _Float16* __restrict__ src, int C, int S,
                              int total /*B*S*C*/, _Float16* __restrict__ out) {
  int tid = blockIdx.x * blockDim.x + threadIdx.x;
  if (tid >= total) return;
  int c = tid % C;
  int b = tid / (S * C);
  out[tid] = src[(size_t)b * C + c];
}

// ---------------------------------------------------------------------------
// 3-NN (top_k(-d,3)) with inverse-distance weights.  Grid = (chunks, B);
// per-batch centers staged in LDS once per block.
// ---------------------------------------------------------------------------
__global__ void knn3(const float* __restrict__ xyz1, const float* __restrict__ xyz2,
                     int N1, int S2, int* __restrict__ kidx, float* __restrict__ kw) {
  __shared__ float sY[360 * 3];
  const int b = blockIdx.y;
  for (int i = threadIdx.x; i < S2 * 3; i += blockDim.x)
    sY[i] = xyz2[(size_t)b * S2 * 3 + i];
  __syncthreads();
  int n = blockIdx.x * blockDim.x + threadIdx.x;
  if (n >= N1) return;
  int row = b * N1 + n;
  float px = xyz1[(size_t)row * 3 + 0], py = xyz1[(size_t)row * 3 + 1], pz = xyz1[(size_t)row * 3 + 2];
  float d0 = 1e30f, d1 = 1e30f, d2 = 1e30f;
  int i0 = 0, i1 = 0, i2 = 0;
  for (int s = 0; s < S2; ++s) {
    float dx = sY[s * 3 + 0] - px, dy = sY[s * 3 + 1] - py, dz = sY[s * 3 + 2] - pz;
    float d = dx * dx + dy * dy + dz * dz;
    if (d < d0)      { d2 = d1; i2 = i1; d1 = d0; i1 = i0; d0 = d; i0 = s; }
    else if (d < d1) { d2 = d1; i2 = i1; d1 = d; i1 = s; }
    else if (d < d2) { d2 = d; i2 = s; }
  }
  float w0 = 1.0f / (d0 + 1e-8f), w1 = 1.0f / (d1 + 1e-8f), w2 = 1.0f / (d2 + 1e-8f);
  float inv = 1.0f / (w0 + w1 + w2);
  kidx[(size_t)row * 3 + 0] = i0; kidx[(size_t)row * 3 + 1] = i1; kidx[(size_t)row * 3 + 2] = i2;
  kw[(size_t)row * 3 + 0] = w0 * inv; kw[(size_t)row * 3 + 1] = w1 * inv; kw[(size_t)row * 3 + 2] = w2 * inv;
}

// interp[row, c] = sum_i kw[row,i] * pts2[b, kidx[row,i], c]
__global__ void interp_apply(const _Float16* __restrict__ pts2, const int* __restrict__ kidx,
                             const float* __restrict__ kw, int S2, int C2, int N1,
                             int total /*B*N1*C2*/, _Float16* __restrict__ out) {
  int tid = blockIdx.x * blockDim.x + threadIdx.x;
  if (tid >= total) return;
  int row = tid / C2, c = tid % C2;
  int b = row / N1;
  float acc = 0.0f;
#pragma unroll
  for (int i = 0; i < 3; ++i) {
    int s = kidx[(size_t)row * 3 + i];
    acc += kw[(size_t)row * 3 + i] * (float)pts2[(size_t)(b * S2 + s) * C2 + c];
  }
  out[tid] = (_Float16)acc;
}

// ---------------------------------------------------------------------------
// Attention core: grid.x = B*NH, block = 128 (one thread per query row).
// Scores row in dynamic LDS (128*128 floats); softmax + V reduction.
// ---------------------------------------------------------------------------
__global__ void attn_core(const _Float16* __restrict__ Q, const _Float16* __restrict__ K,
                          const _Float16* __restrict__ V, _Float16* __restrict__ O) {
  extern __shared__ float sc[];
  const int S = 128, D = 64, C = 256;
  const int bh = blockIdx.x;
  const int b = bh >> 2, h = bh & 3;
  const int s = threadIdx.x;
  const _Float16* q = Q + (size_t)(b * S + s) * C + h * D;
  float* row = sc + (size_t)s * S;
  float qr[64];
#pragma unroll
  for (int d = 0; d < D; ++d) qr[d] = (float)q[d];
  float mx = -1e30f;
  for (int t = 0; t < S; ++t) {
    const _Float16* k = K + (size_t)(b * S + t) * C + h * D;
    float acc = 0.0f;
    for (int d = 0; d < D; ++d) acc += qr[d] * (float)k[d];
    acc *= 0.125f;  // 1/sqrt(64)
    row[t] = acc;
    mx = fmaxf(mx, acc);
  }
  float sum = 0.0f;
  for (int t = 0; t < S; ++t) { float e = __expf(row[t] - mx); row[t] = e; sum += e; }
  float inv = 1.0f / sum;
  for (int d = 0; d < D; ++d) {
    float acc = 0.0f;
    for (int t = 0; t < S; ++t)
      acc += row[t] * (float)V[(size_t)(b * S + t) * C + h * D + d];
    O[(size_t)(b * S + s) * C + h * D + d] = (_Float16)(acc * inv);
  }
}

// out = layernorm(a + b), C = 256, block = 256 (one row per block)
__global__ void add_layernorm(const _Float16* __restrict__ a, const _Float16* __restrict__ b,
                              _Float16* __restrict__ out) {
  __shared__ float red[256];
  const int C = 256;
  const int row = blockIdx.x, c = threadIdx.x;
  float v = (float)a[(size_t)row * C + c] + (float)b[(size_t)row * C + c];
  red[c] = v;
  __syncthreads();
  for (int s = 128; s > 0; s >>= 1) { if (c < s) red[c] += red[c + s]; __syncthreads(); }
  const float mean = red[0] / (float)C;
  __syncthreads();
  float d = v - mean;
  red[c] = d * d;
  __syncthreads();
  for (int s = 128; s > 0; s >>= 1) { if (c < s) red[c] += red[c + s]; __syncthreads(); }
  const float var = red[0] / (float)C;
  out[(size_t)row * C + c] = (_Float16)(d * rsqrtf(var + 1e-5f));
}

// ===========================================================================
// Host orchestration
// ===========================================================================
extern "C" void kernel_launch(void* const* d_in, const int* in_sizes, int n_in,
                              void* d_out, int out_size, void* d_ws, size_t ws_size,
                              hipStream_t stream) {
  (void)in_sizes; (void)n_in; (void)out_size; (void)ws_size;
  const int B = 16, N0 = 8192, NP1 = 358, NP2 = 128;

  // ---- workspace carving -------------------------------------------------
  char* base = (char*)d_ws;
  size_t off = 0;
  auto alloc = [&](size_t bytes) -> char* {
    size_t a = (off + 255) & ~(size_t)255;
    off = a + bytes;
    return base + a;
  };

  // ---- input leaf table (insertion order of setup_inputs) ----------------
  const float* x = (const float*)d_in[0];
  struct WDesc { int idx, K, N; };
  const WDesc wd[] = {
      {1, 6, 32},   {3, 32, 32},   {5, 32, 64},
      {7, 6, 64},   {9, 64, 64},   {11, 64, 128},
      {13, 6, 64},  {15, 64, 96},  {17, 96, 128},
      {19, 323, 128},{21, 128, 128},{23, 128, 256},
      {25, 323, 128},{27, 128, 196},{29, 196, 256},
      {31, 515, 256},{33, 256, 512},{35, 512, 1024},
      {37, 256, 256},{38, 256, 256},{39, 256, 256},{40, 256, 256},
      {41, 256, 1024},{42, 1024, 256},
      {43, 1536, 256},{45, 256, 256},
      {47, 576, 256},{49, 256, 128},
      {51, 134, 128},{53, 128, 256}};
  const int nW = (int)(sizeof(wd) / sizeof(wd[0]));
  _Float16* w16[64] = {};
  for (int i = 0; i < nW; ++i) {
    int Kt = (wd[i].K + 31) / 32;
    int Nt = round32(wd[i].N) / 16;
    int total = Kt * Nt * 512;
    _Float16* p = (_Float16*)alloc((size_t)total * 2);
    w16[wd[i].idx] = p;
    pack_weights<<<(total + 255) / 256, 256, 0, stream>>>(
        (const float*)d_in[wd[i].idx], wd[i].K, wd[i].N, Kt, Nt, p);
  }

  // ---- big scratch buffers ----------------------------------------------
  const size_t PP = (size_t)B * NP1 * 128 * 128 + 256 * 1536;  // headroom for M pad
  _Float16* ping = (_Float16*)alloc(PP * 2);
  _Float16* pong = (_Float16*)alloc(PP * 2);
  int* idx1 = (int*)alloc((size_t)B * NP1 * 4);
  float* l1_xyz = (float*)alloc((size_t)B * NP1 * 3 * 4);
  int* gi = (int*)alloc((size_t)B * NP1 * 128 * 4);
  _Float16* l1_points = (_Float16*)alloc((size_t)B * NP1 * 320 * 2);
  int* idx2 = (int*)alloc((size_t)B * NP2 * 4);
  float* l2_xyz = (float*)alloc((size_t)B * NP2 * 3 * 4);
  _Float16* l2p512 = (_Float16*)alloc((size_t)B * NP2 * 512 * 2);
  _Float16* l3 = (_Float16*)alloc((size_t)B * 1024 * 2);
  _Float16* interp = (_Float16*)alloc((size_t)B * N0 * 128 * 2);
  int* kidx = (int*)alloc((size_t)B * N0 * 3 * 4);
  float* kw = (float*)alloc((size_t)B * N0 * 3 * 4);
  _Float16* xq = (_Float16*)alloc((size_t)B * NP2 * 256 * 2);
  _Float16* Qb = (_Float16*)alloc((size_t)B * NP2 * 256 * 2);
  _Float16* Kb = (_Float16*)alloc((size_t)B * NP2 * 256 * 2);
  _Float16* Vb = (_Float16*)alloc((size_t)B * NP2 * 256 * 2);
  _Float16* Ob = (_Float16*)alloc((size_t)B * NP2 * 256 * 2);
  _Float16* oW = (_Float16*)alloc((size_t)B * NP2 * 256 * 2);
  _Float16* attx = (_Float16*)alloc((size_t)B * NP2 * 256 * 2);
  _Float16* ffb = (_Float16*)alloc((size_t)B * NP2 * 1024 * 2);
  _Float16* ff2o = (_Float16*)alloc((size_t)B * NP2 * 256 * 2);
  _Float16* l2p256 = (_Float16*)alloc((size_t)B * NP2 * 256 * 2);
  _Float16* l1up = (_Float16*)alloc((size_t)(B * NP1 + 128) * 128 * 2);

  // ---- GEMM launcher -----------------------------------------------------
  auto gemm = [&](const _Float16* A, int leafW, void* out, int M, int K, int N,
                  bool relu, bool f32out, bool useBias) {
    const _Float16* W = w16[leafW];
    const float* bias = useBias ? (const float*)d_in[leafW + 1] : nullptr;
    int Kt = (K + 31) / 32;
    int lda = Kt * 32;
    int ldo = f32out ? N : round32(N);
    int Nt = round32(N) / 16;
    dim3 g((unsigned)((M + 127) / 128), (unsigned)Nt);
    if (relu) {
      if (f32out) gemm_wmma<true, true><<<g, 256, 0, stream>>>(A, lda, W, bias, out, M, Kt, Nt, N, ldo);
      else        gemm_wmma<true, false><<<g, 256, 0, stream>>>(A, lda, W, bias, out, M, Kt, Nt, N, ldo);
    } else {
      if (f32out) gemm_wmma<false, true><<<g, 256, 0, stream>>>(A, lda, W, bias, out, M, Kt, Nt, N, ldo);
      else        gemm_wmma<false, false><<<g, 256, 0, stream>>>(A, lda, W, bias, out, M, Kt, Nt, N, ldo);
    }
  };

  // =========================================================================
  // SA level 1 (MSG): FPS 8192 -> 358  (xyz + dist fully in LDS, 1024 thr)
  // =========================================================================
  {
    size_t sm = (size_t)N0 * 16 + 1024 * 8 + 16;
    fps_kernel<<<B, 1024, sm, stream>>>(x, N0, NP1, idx1);
    int tot = B * NP1;
    gather_xyz<<<(tot * 3 + 255) / 256, 256, 0, stream>>>(x, idx1, N0, NP1, tot, l1_xyz);
  }
  struct Branch { float r; int ns; int w[3]; int wi[4]; int coff; };
  const Branch ds1[3] = {
      {0.1f, 32, {1, 3, 5},   {6, 32, 32, 64},  0},
      {0.2f, 64, {7, 9, 11},  {6, 64, 64, 128}, 64},
      {0.4f, 128,{13, 15, 17},{6, 64, 96, 128}, 192}};
  for (int br = 0; br < 3; ++br) {
    const Branch& bd = ds1[br];
    int tot = B * NP1;
    query_ball<<<(tot + 7) / 8, 256, 0, stream>>>(x, l1_xyz, N0, NP1, bd.r * bd.r, bd.ns, gi, tot);
    int rows = tot * bd.ns;
    int Cpad = round32(bd.wi[0]);  // 6 -> 32
    size_t totalE = (size_t)rows * Cpad;
    group_points<float><<<(unsigned)((totalE + 255) / 256), 256, 0, stream>>>(
        x, l1_xyz, x, 3, gi, N0, NP1, bd.ns, rows, Cpad, ping);
    _Float16* cur = ping; _Float16* nxt = pong;
    for (int l = 0; l < 3; ++l) {
      gemm(cur, bd.w[l], nxt, rows, bd.wi[l], bd.wi[l + 1], true, false, true);
      _Float16* t = cur; cur = nxt; nxt = t;
    }
    int C = bd.wi[3];
    maxpool_rows<<<(tot * C + 255) / 256, 256, 0, stream>>>(cur, tot, bd.ns, C,
                                                            l1_points, 320, bd.coff);
  }

  // =========================================================================
  // SA level 2 (MSG): FPS 358 -> 128
  // =========================================================================
  {
    size_t sm = (size_t)NP1 * 16 + 256 * 8 + 16;
    fps_kernel<<<B, 256, sm, stream>>>(l1_xyz, NP1, NP2, idx2);
    int tot = B * NP2;
    gather_xyz<<<(tot * 3 + 255) / 256, 256, 0, stream>>>(l1_xyz, idx2, NP1, NP2, tot, l2_xyz);
  }
  const Branch ds2[2] = {
      {0.4f, 64, {19, 21, 23}, {323, 128, 128, 256}, 0},
      {0.8f, 128,{25, 27, 29}, {323, 128, 196, 256}, 256}};
  for (int br = 0; br < 2; ++br) {
    const Branch& bd = ds2[br];
    int tot = B * NP2;
    query_ball<<<(tot + 7) / 8, 256, 0, stream>>>(l1_xyz, l2_xyz, NP1, NP2, bd.r * bd.r, bd.ns, gi, tot);
    int rows = tot * bd.ns;
    int Cpad = round32(bd.wi[0]);  // 323 -> 352
    size_t totalE = (size_t)rows * Cpad;
    group_points<_Float16><<<(unsigned)((totalE + 255) / 256), 256, 0, stream>>>(
        l1_xyz, l2_xyz, l1_points, 320, gi, NP1, NP2, bd.ns, rows, Cpad, ping);
    _Float16* cur = ping; _Float16* nxt = pong;
    for (int l = 0; l < 3; ++l) {
      gemm(cur, bd.w[l], nxt, rows, bd.wi[l], bd.wi[l + 1], true, false, true);
      _Float16* t = cur; cur = nxt; nxt = t;
    }
    int C = bd.wi[3];
    maxpool_rows<<<(tot * C + 255) / 256, 256, 0, stream>>>(cur, tot, bd.ns, C,
                                                            l2p512, 512, bd.coff);
  }

  // =========================================================================
  // SA level 3 (group all): [515 -> 256 -> 512 -> 1024], max over 128 points
  // =========================================================================
  {
    int tot = B * NP2;
    int Cpad = round32(515);  // 544
    pack_groupall<<<(tot * Cpad + 255) / 256, 256, 0, stream>>>(l2_xyz, l2p512, 512, tot, Cpad, ping);
    gemm(ping, 31, pong, tot, 515, 256, true, false, true);
    gemm(pong, 33, ping, tot, 256, 512, true, false, true);
    gemm(ping, 35, pong, tot, 512, 1024, true, false, true);
    maxpool_rows<<<(B * 1024 + 255) / 256, 256, 0, stream>>>(pong, B, NP2, 1024, l3, 1024, 0);
  }

  // =========================================================================
  // FP up3: broadcast l3 (S==1), concat [l2p512 | l3bc] -> [1536->256->256]
  // =========================================================================
  {
    int tot = B * NP2;
    int n = tot * 1024;
    broadcast_pts<<<(n + 255) / 256, 256, 0, stream>>>(l3, 1024, NP2, n, interp);
    pack_concat_f16<<<(tot * 1536 + 255) / 256, 256, 0, stream>>>(l2p512, 512, interp, 1024, tot, 1536, ping);
    gemm(ping, 43, pong, tot, 1536, 256, true, false, true);
    gemm(pong, 45, xq, tot, 256, 256, true, false, true);
  }

  // =========================================================================
  // Attention block on (B, 128, 256), 4 heads
  // =========================================================================
  {
    int M = B * NP2;
    gemm(xq, 37, Qb, M, 256, 256, false, false, false);
    gemm(xq, 38, Kb, M, 256, 256, false, false, false);
    gemm(xq, 39, Vb, M, 256, 256, false, false, false);
    attn_core<<<B * 4, 128, 128 * 128 * 4, stream>>>(Qb, Kb, Vb, Ob);
    gemm(Ob, 40, oW, M, 256, 256, false, false, false);
    add_layernorm<<<M, 256, 0, stream>>>(xq, oW, attx);
    gemm(attx, 41, ffb, M, 256, 1024, true, false, false);
    gemm(ffb, 42, ff2o, M, 1024, 256, false, false, false);
    add_layernorm<<<M, 256, 0, stream>>>(attx, ff2o, l2p256);
  }

  // =========================================================================
  // FP up2: 3-NN interp l2p256 -> 358 pts, concat with l1_points (320) ->
  //         [576 -> 256 -> 128]
  // =========================================================================
  {
    int tot = B * NP1;
    knn3<<<dim3((NP1 + 255) / 256, B), 256, 0, stream>>>(l1_xyz, l2_xyz, NP1, NP2, kidx, kw);
    int n = tot * 256;
    interp_apply<<<(n + 255) / 256, 256, 0, stream>>>(l2p256, kidx, kw, NP2, 256, NP1, n, interp);
    pack_concat_f16<<<(tot * 576 + 255) / 256, 256, 0, stream>>>(l1_points, 320, interp, 256, tot, 576, ping);
    gemm(ping, 47, pong, tot, 576, 256, true, false, true);
    gemm(pong, 49, l1up, tot, 256, 128, true, false, true);
  }

  // =========================================================================
  // FP up1: 3-NN interp l1up -> 8192 pts, concat [x|x|interp] = 134 -> 160pad
  //         [134 -> 128 -> 256], final layer fp32 into d_out
  // =========================================================================
  {
    int tot = B * N0;
    knn3<<<dim3((N0 + 255) / 256, B), 256, 0, stream>>>(x, l1_xyz, N0, NP1, kidx, kw);
    int n = tot * 128;
    interp_apply<<<(n + 255) / 256, 256, 0, stream>>>(l1up, kidx, kw, NP1, 128, N0, n, interp);
    pack_up1<<<(tot * 160 + 255) / 256, 256, 0, stream>>>(x, interp, tot, ping);
    gemm(ping, 51, pong, tot, 134, 128, true, false, true);
    gemm(pong, 53, d_out, tot, 128, 256, true, true, true);
  }
}
